// TimeTermGCN_60017872995138
// MI455X (gfx1250) — matrix operand order repaired and checked by reading, not compiled, not run on hardware
//
#include <hip/hip_runtime.h>
#include <hip/hip_bf16.h>
#include <math.h>

// ---------------------------------------------------------------------------
// CDNA5 (gfx1250) WMMA pipeline for TimeTermGCN forward.
// bf16 WMMA 16x16x32 for GEMMs, f32 WMMA 16x16x4 for attention logits.
// ---------------------------------------------------------------------------

typedef __attribute__((ext_vector_type(16))) __bf16 v16bf;
typedef __attribute__((ext_vector_type(8)))  float  v8f;
typedef __attribute__((ext_vector_type(2)))  float  v2f;

union FragBF {
    v16bf v;
    uint4 q[2];
};

__device__ inline v16bf frag_zero() {
    FragBF f;
    f.q[0] = make_uint4(0u, 0u, 0u, 0u);
    f.q[1] = make_uint4(0u, 0u, 0u, 0u);
    return f.v;
}

// A fragment (16x32 bf16), row-major source [M][K].
// element e of lane L: M = L%16, K = kc + 16*(L/16) + e
__device__ inline v16bf load_frag_a(const __bf16* __restrict__ A, int lda,
                                    int row0, int kc, int lane) {
    const int m = lane & 15, h = lane >> 4;
    const __bf16* base = A + (size_t)(row0 + m) * (size_t)lda + kc;
    FragBF f;
    f.q[0] = *(const uint4*)(base + 16 * h);       // e=0..7  -> K = kc+16h+e
    f.q[1] = *(const uint4*)(base + 16 * h + 8);   // e=8..15 -> K = kc+16h+e
    return f.v;
}

// B fragment (32x16 bf16), source is B^T row-major [N][K].
// element e of lane L: N = L%16, K = kc + 16*(L/16) + e  (contiguous 32B)
__device__ inline v16bf load_frag_b(const __bf16* __restrict__ Bt, int ldb,
                                    int col0, int kc, int lane) {
    const int n = lane & 15, h = lane >> 4;
    const uint4* p = (const uint4*)(Bt + (size_t)(col0 + n) * (size_t)ldb + kc + 16 * h);
    FragBF f;
    f.q[0] = p[0];
    f.q[1] = p[1];
    return f.v;
}

__device__ inline void zero_acc(v8f& a) {
#pragma unroll
    for (int e = 0; e < 8; ++e) a[e] = 0.0f;
}

// ---------------------------------------------------------------------------
// Generic batched GEMM: C[b] = A[b] (MxK, row-major) * B[b] (Bt = [N][K]) (+bias)
// biasMode: 0 = none, 1 = bias[n], 2 = bias[m]
// Block tile 128x64, 8 waves of 32x32 each.
// ---------------------------------------------------------------------------
__global__ __launch_bounds__(256) void gemm_bf16_wmma(
    const __bf16* __restrict__ A, const __bf16* __restrict__ Bt,
    __bf16* __restrict__ C, const float* __restrict__ bias,
    int M, int N, int K, int biasMode,
    unsigned long long sA, unsigned long long sB, unsigned long long sC)
{
    const int b = blockIdx.x;
    const __bf16* Ab = A + (size_t)b * sA;
    const __bf16* Bb = Bt + (size_t)b * sB;
    __bf16* Cb = C + (size_t)b * sC;
    const int lane = threadIdx.x & 31;
    const int w = threadIdx.x >> 5;
    const int r0 = blockIdx.y * 128 + (w >> 1) * 32;
    const int c0 = blockIdx.z * 64 + (w & 1) * 32;
    if (r0 >= M || c0 >= N) return;

    v8f acc[2][2];
    zero_acc(acc[0][0]); zero_acc(acc[0][1]);
    zero_acc(acc[1][0]); zero_acc(acc[1][1]);

    for (int kc = 0; kc < K; kc += 32) {
        v16bf a0 = load_frag_a(Ab, K, r0, kc, lane);
        v16bf a1 = load_frag_a(Ab, K, r0 + 16, kc, lane);
        v16bf b0 = load_frag_b(Bb, K, c0, kc, lane);
        v16bf b1 = load_frag_b(Bb, K, c0 + 16, kc, lane);
        acc[0][0] = __builtin_amdgcn_wmma_f32_16x16x32_bf16(false, a0, false, b0, (short)0, acc[0][0], false, false);
        acc[0][1] = __builtin_amdgcn_wmma_f32_16x16x32_bf16(false, a0, false, b1, (short)0, acc[0][1], false, false);
        acc[1][0] = __builtin_amdgcn_wmma_f32_16x16x32_bf16(false, a1, false, b0, (short)0, acc[1][0], false, false);
        acc[1][1] = __builtin_amdgcn_wmma_f32_16x16x32_bf16(false, a1, false, b1, (short)0, acc[1][1], false, false);
    }

    const int n_in = lane & 15, h = lane >> 4;
#pragma unroll
    for (int i = 0; i < 2; ++i)
#pragma unroll
        for (int j = 0; j < 2; ++j) {
            const int ng = c0 + j * 16 + n_in;
            float bn = (biasMode == 1) ? bias[ng] : 0.0f;
#pragma unroll
            for (int e = 0; e < 8; ++e) {
                const int mg = r0 + i * 16 + e + 8 * h;
                float v = acc[i][j][e] + bn;
                if (biasMode == 2) v += bias[mg];
                Cb[(size_t)mg * (size_t)N + ng] = (__bf16)v;
            }
        }
}

// ---------------------------------------------------------------------------
// Causal dilated conv as 3 accumulated GEMMs:
//   Y[b][o][l] = sum_k sum_i Wp[k][o][i] * XfT[b][l-4+2k][i] + bias[o]
// Wp: [3][128][128] bf16, XfT: [B][256][128] bf16 (= X_feat transposed)
// ---------------------------------------------------------------------------
__global__ __launch_bounds__(256) void conv_wmma(
    const __bf16* __restrict__ Wp, const __bf16* __restrict__ XfT,
    __bf16* __restrict__ Y, const float* __restrict__ bias)
{
    const int b = blockIdx.x;
    const int lane = threadIdx.x & 31;
    const int w = threadIdx.x >> 5;
    const int r0 = (w >> 1) * 32;                    // M = 128
    const int c0 = blockIdx.z * 64 + (w & 1) * 32;   // N = 256
    const __bf16* Xb = XfT + (size_t)b * 256 * 128;
    const int n = lane & 15;

    v8f acc[2][2];
    zero_acc(acc[0][0]); zero_acc(acc[0][1]);
    zero_acc(acc[1][0]); zero_acc(acc[1][1]);

    for (int tap = 0; tap < 3; ++tap) {
        const __bf16* Ak = Wp + tap * 128 * 128;
        const int shift = 2 * tap - 4;   // source column = l + shift
        for (int kc = 0; kc < 128; kc += 32) {
            v16bf a0 = load_frag_a(Ak, 128, r0, kc, lane);
            v16bf a1 = load_frag_a(Ak, 128, r0 + 16, kc, lane);
            v16bf b0 = ((c0 + n + shift) >= 0)
                           ? load_frag_b(Xb, 128, c0 + shift, kc, lane) : frag_zero();
            v16bf b1 = ((c0 + 16 + n + shift) >= 0)
                           ? load_frag_b(Xb, 128, c0 + 16 + shift, kc, lane) : frag_zero();
            acc[0][0] = __builtin_amdgcn_wmma_f32_16x16x32_bf16(false, a0, false, b0, (short)0, acc[0][0], false, false);
            acc[0][1] = __builtin_amdgcn_wmma_f32_16x16x32_bf16(false, a0, false, b1, (short)0, acc[0][1], false, false);
            acc[1][0] = __builtin_amdgcn_wmma_f32_16x16x32_bf16(false, a1, false, b0, (short)0, acc[1][0], false, false);
            acc[1][1] = __builtin_amdgcn_wmma_f32_16x16x32_bf16(false, a1, false, b1, (short)0, acc[1][1], false, false);
        }
    }

    const int h = lane >> 4;
#pragma unroll
    for (int i = 0; i < 2; ++i)
#pragma unroll
        for (int j = 0; j < 2; ++j) {
            const int ng = c0 + j * 16 + n;
#pragma unroll
            for (int e = 0; e < 8; ++e) {
                const int mg = r0 + i * 16 + e + 8 * h;
                float v = acc[i][j][e] + bias[mg];
                Y[((size_t)b * 128 + mg) * 256 + ng] = (__bf16)v;
            }
        }
}

// ---------------------------------------------------------------------------
// Attention: logits via f32 WMMA 16x16x4, fused softmax, atomic batch-mean.
// X: [B][nodes][d] (f32 or bf16). Block handles a full 16-row slab.
// NF = nodes/128 column-fragments per wave (1 for feat, 2 for time).
// ---------------------------------------------------------------------------
template <int NF, int ISBF16>
__global__ __launch_bounds__(256) void attn_softmax_wmma(
    const void* __restrict__ Xv,
    const float* __restrict__ wq, const float* __restrict__ wk,
    float* __restrict__ accum, int nodes, int d, float scale, float invB)
{
    __shared__ float sm[16 * 256];
    __shared__ float rmax[16];
    __shared__ float rinv[16];

    const int b = blockIdx.x;
    const int r0 = blockIdx.y * 16;
    const int tid = threadIdx.x;
    const int lane = tid & 31;
    const int w = tid >> 5;
    const int m = lane & 15, h = lane >> 4;
    const float* Xf = (const float*)Xv;
    const __bf16* Xb = (const __bf16*)Xv;
    const size_t bbase = (size_t)b * (size_t)nodes * (size_t)d;

    v8f acc[NF];
#pragma unroll
    for (int j = 0; j < NF; ++j) zero_acc(acc[j]);

    for (int kc = 0; kc < d; kc += 4) {
        const int ka = kc + 2 * h;
        v2f a;
        if (ISBF16) {
            const __bf16* p = Xb + bbase + (size_t)(r0 + m) * d + ka;
            a.x = (float)p[0]; a.y = (float)p[1];
        } else {
            a = *(const v2f*)(Xf + bbase + (size_t)(r0 + m) * d + ka);
        }
#pragma unroll
        for (int j = 0; j < NF; ++j) {
            const int cg = w * (16 * NF) + j * 16 + m;
            v2f bb;
            if (ISBF16) {
                const __bf16* p = Xb + bbase + (size_t)cg * d + ka;
                bb.x = (float)p[0]; bb.y = (float)p[1];
            } else {
                bb = *(const v2f*)(Xf + bbase + (size_t)cg * d + ka);
            }
            acc[j] = __builtin_amdgcn_wmma_f32_16x16x4_f32(false, a, false, bb, (short)0, acc[j], false, false);
        }
    }

    // scale by diag(wq)*diag(wk)/sqrt(d) and stage into LDS
#pragma unroll
    for (int j = 0; j < NF; ++j) {
        const int cl = w * (16 * NF) + j * 16 + m;
        const float dk = wk[(size_t)cl * nodes + cl];
#pragma unroll
        for (int e = 0; e < 8; ++e) {
            const int mr = e + 8 * h;
            const int mg = r0 + mr;
            const float dq = wq[(size_t)mg * nodes + mg];
            sm[mr * nodes + cl] = acc[j][e] * dq * dk * scale;
        }
    }
    __syncthreads();

    if (tid < 16) {
        float mx = -1e30f;
        for (int c = 0; c < nodes; ++c) mx = fmaxf(mx, sm[tid * nodes + c]);
        rmax[tid] = mx;
    }
    __syncthreads();
    for (int idx = tid; idx < 16 * nodes; idx += 256)
        sm[idx] = __expf(sm[idx] - rmax[idx / nodes]);
    __syncthreads();
    if (tid < 16) {
        float s = 0.0f;
        for (int c = 0; c < nodes; ++c) s += sm[tid * nodes + c];
        rinv[tid] = 1.0f / s;
    }
    __syncthreads();
    for (int idx = tid; idx < 16 * nodes; idx += 256) {
        const int r = idx / nodes;
        const int c = idx - r * nodes;
        atomicAdd(&accum[(size_t)(r0 + r) * nodes + c], sm[idx] * rinv[r] * invB);
    }
}

// ---------------------------------------------------------------------------
// GCN normalization: Mt[c][r] = dis[r]*A[r][c]*dis[c]  (bf16 out, single block)
// ---------------------------------------------------------------------------
__global__ __launch_bounds__(256) void gcn_norm(const float* __restrict__ A,
                                                __bf16* __restrict__ Mt, int nodes)
{
    __shared__ float dis[256];
    const int t = threadIdx.x;
    if (t < nodes) {
        float s = 0.0f;
        for (int r = 0; r < nodes; ++r) s += A[(size_t)r * nodes + t];
        dis[t] = (s > 0.0f) ? rsqrtf(s) : 0.0f;
    }
    __syncthreads();
    for (int idx = t; idx < nodes * nodes; idx += 256) {
        const int r = idx / nodes;
        const int c = idx - r * nodes;
        Mt[(size_t)c * nodes + r] = (__bf16)(dis[r] * A[idx] * dis[c]);
    }
}

// ---------------------------------------------------------------------------
// Small utility kernels
// ---------------------------------------------------------------------------
__global__ void cvt_f2b(const float* __restrict__ s, __bf16* __restrict__ d, long long n)
{
    for (long long i = (long long)blockIdx.x * 256 + threadIdx.x; i < n;
         i += (long long)gridDim.x * 256)
        d[i] = (__bf16)s[i];
}

// dst[b][c][r] = src[b][r][c]  (f32 -> bf16)
__global__ void transpose_f2b(const float* __restrict__ src, __bf16* __restrict__ dst,
                              int R, int C, long long total)
{
    const long long rc = (long long)R * C;
    for (long long idx = (long long)blockIdx.x * 256 + threadIdx.x; idx < total;
         idx += (long long)gridDim.x * 256) {
        const long long bb = idx / rc;
        const long long rem = idx - bb * rc;
        const int r = (int)(rem / C);
        const int c = (int)(rem - (long long)r * C);
        dst[bb * rc + (long long)c * R + r] = (__bf16)src[idx];
    }
}

// dst[b][c][r] = src[b][r][c]  (bf16 -> bf16)
__global__ void transpose_b2b(const __bf16* __restrict__ src, __bf16* __restrict__ dst,
                              int R, int C, long long total)
{
    const long long rc = (long long)R * C;
    for (long long idx = (long long)blockIdx.x * 256 + threadIdx.x; idx < total;
         idx += (long long)gridDim.x * 256) {
        const long long bb = idx / rc;
        const long long rem = idx - bb * rc;
        const int r = (int)(rem / C);
        const int c = (int)(rem - (long long)r * C);
        dst[bb * rc + (long long)c * R + r] = src[idx];
    }
}

// conv_w [128][128][3] f32 -> Wp [3][128][128] bf16
__global__ void pack_conv(const float* __restrict__ w, __bf16* __restrict__ out)
{
    const int idx = blockIdx.x * 256 + threadIdx.x;
    if (idx >= 128 * 128 * 3) return;
    const int o = idx / (128 * 3);
    const int rem = idx - o * (128 * 3);
    const int i = rem / 3;
    const int k = rem - i * 3;
    out[k * 128 * 128 + o * 128 + i] = (__bf16)w[idx];
}

// out[b][h] = mean_c h[b][c][h]   (h bf16 [B][C][H], H == blockDim.x)
__global__ void mean_nodes(const __bf16* __restrict__ hsrc, float* __restrict__ out,
                           int C, int H)
{
    const int b = blockIdx.x;
    const int hh = threadIdx.x;
    float s = 0.0f;
    for (int c = 0; c < C; ++c)
        s += (float)hsrc[((size_t)b * C + c) * H + hh];
    out[(size_t)b * H + hh] = s / (float)C;
}

// out[b][o] = [fm|tm][b] @ proj_w + proj_b
__global__ void proj_kernel(const float* __restrict__ fm, const float* __restrict__ tm,
                            const float* __restrict__ pw, const float* __restrict__ pb,
                            float* __restrict__ out)
{
    const int b = blockIdx.x;
    const int o = threadIdx.x;   // 64
    float s = pb[o];
    for (int h = 0; h < 128; ++h) s += fm[b * 128 + h] * pw[h * 64 + o];
    for (int h = 0; h < 128; ++h) s += tm[b * 128 + h] * pw[(128 + h) * 64 + o];
    out[b * 64 + o] = s;
}

// ---------------------------------------------------------------------------
// Host orchestration
// ---------------------------------------------------------------------------
extern "C" void kernel_launch(void* const* d_in, const int* in_sizes, int n_in,
                              void* d_out, int out_size, void* d_ws, size_t ws_size,
                              hipStream_t stream)
{
    constexpr int B = 512, Wd = 256, F = 128, H = 128;
    const float* X_time = (const float*)d_in[0];
    const float* conv_w = (const float*)d_in[1];
    const float* conv_b = (const float*)d_in[2];
    const float* wq_feat = (const float*)d_in[3];
    const float* wk_feat = (const float*)d_in[4];
    const float* wq_time = (const float*)d_in[5];
    const float* wk_time = (const float*)d_in[6];
    const float* gf_w[3] = { (const float*)d_in[7], (const float*)d_in[9], (const float*)d_in[11] };
    const float* gf_b[3] = { (const float*)d_in[8], (const float*)d_in[10], (const float*)d_in[12] };
    const float* gt_w[3] = { (const float*)d_in[13], (const float*)d_in[15], (const float*)d_in[17] };
    const float* gt_b[3] = { (const float*)d_in[14], (const float*)d_in[16], (const float*)d_in[18] };
    const float* proj_w = (const float*)d_in[19];
    const float* proj_b = (const float*)d_in[20];

    float* outF = (float*)d_out;
    float* Afeat = outF + (size_t)B * 64;                 // [128][128]
    float* Atime = Afeat + (size_t)F * F;                 // [256][256]

    // workspace layout
    char* ws = (char*)d_ws;
    size_t off = 0;
    auto alloc = [&](size_t bytes) {
        void* p = ws + off;
        off = (off + bytes + 255) & ~(size_t)255;
        return p;
    };
    const size_t big = (size_t)B * Wd * F * sizeof(__bf16);   // 33.5 MB
    __bf16* Xt_bf = (__bf16*)alloc(big);      // X_time bf16 [B][256][128]
    __bf16* Xc_bf = (__bf16*)alloc(big);      // conv out  [B][128][256]
    __bf16* S1 = (__bf16*)alloc(big);
    __bf16* S2 = (__bf16*)alloc(big);         // also used as XfT [B][256][128]
    __bf16* S3 = (__bf16*)alloc(big);
    __bf16* wtF[3], * wtT[3];
    for (int i = 0; i < 3; ++i) wtF[i] = (__bf16*)alloc(128 * 256 * sizeof(__bf16));
    for (int i = 0; i < 3; ++i) wtT[i] = (__bf16*)alloc(128 * 128 * sizeof(__bf16));
    __bf16* Wp = (__bf16*)alloc(3 * 128 * 128 * sizeof(__bf16));
    __bf16* Mtf = (__bf16*)alloc(128 * 128 * sizeof(__bf16));
    __bf16* Mtt = (__bf16*)alloc(256 * 256 * sizeof(__bf16));
    float* fm = (float*)alloc((size_t)B * H * sizeof(float));
    float* tm = (float*)alloc((size_t)B * H * sizeof(float));
    (void)ws_size; (void)in_sizes; (void)n_in; (void)out_size;

    const long long nX = (long long)B * Wd * F;
    const int tb = 256;
    auto blocks = [](long long n) { return (unsigned)((n + 255) / 256); };

    // --- conversions / packing ---
    cvt_f2b<<<blocks(nX), tb, 0, stream>>>(X_time, Xt_bf, nX);
    // XfT[b][l][i] = X_feat[b][i][l], X_feat = reshape(X_time, [B,128,256])
    transpose_f2b<<<blocks(nX), tb, 0, stream>>>(X_time, S2, 128, 256, nX);
    pack_conv<<<blocks(128 * 128 * 3), tb, 0, stream>>>(conv_w, Wp);
    transpose_f2b<<<blocks(256 * 128), tb, 0, stream>>>(gf_w[0], wtF[0], 256, 128, 256 * 128);
    transpose_f2b<<<blocks(128 * 128), tb, 0, stream>>>(gf_w[1], wtF[1], 128, 128, 128 * 128);
    transpose_f2b<<<blocks(128 * 128), tb, 0, stream>>>(gf_w[2], wtF[2], 128, 128, 128 * 128);
    for (int i = 0; i < 3; ++i)
        transpose_f2b<<<blocks(128 * 128), tb, 0, stream>>>(gt_w[i], wtT[i], 128, 128, 128 * 128);

    // --- conv (uses S2 = XfT) ---
    conv_wmma<<<dim3(B, 1, 4), tb, 0, stream>>>(Wp, S2, Xc_bf, conv_b);

    // --- zero attention accumulators (they live in d_out) ---
    hipMemsetAsync(Afeat, 0, (size_t)(F * F + Wd * Wd) * sizeof(float), stream);

    // --- feat branch ---
    attn_softmax_wmma<1, 1><<<dim3(B, F / 16), tb, 0, stream>>>(
        Xc_bf, wq_feat, wk_feat, Afeat, F, Wd, 0.0625f, 1.0f / (float)B);
    gcn_norm<<<1, tb, 0, stream>>>(Afeat, Mtf, F);

    const unsigned long long sX = (unsigned long long)Wd * F;   // 32768
    const unsigned long long sH = (unsigned long long)F * H;    // 16384
    // layer 1
    gemm_bf16_wmma<<<dim3(B, 1, 2), tb, 0, stream>>>(Xc_bf, wtF[0], S1, nullptr,
        128, 128, 256, 0, sX, 0ull, sH);
    transpose_b2b<<<blocks((long long)B * sH), tb, 0, stream>>>(S1, S2, 128, 128, (long long)B * sH);
    gemm_bf16_wmma<<<dim3(B, 1, 2), tb, 0, stream>>>(Mtf, S2, S3, gf_b[0],
        128, 128, 128, 1, 0ull, sH, sH);
    // layers 2,3
    for (int l = 1; l < 3; ++l) {
        gemm_bf16_wmma<<<dim3(B, 1, 2), tb, 0, stream>>>(S3, wtF[l], S1, nullptr,
            128, 128, 128, 0, sH, 0ull, sH);
        transpose_b2b<<<blocks((long long)B * sH), tb, 0, stream>>>(S1, S2, 128, 128, (long long)B * sH);
        gemm_bf16_wmma<<<dim3(B, 1, 2), tb, 0, stream>>>(Mtf, S2, S3, gf_b[l],
            128, 128, 128, 1, 0ull, sH, sH);
    }
    mean_nodes<<<B, 128, 0, stream>>>(S3, fm, 128, 128);

    // --- time branch ---
    attn_softmax_wmma<2, 0><<<dim3(B, Wd / 16), tb, 0, stream>>>(
        X_time, wq_time, wk_time, Atime, Wd, F, 0.08838834764831845f, 1.0f / (float)B);
    gcn_norm<<<1, tb, 0, stream>>>(Atime, Mtt, Wd);

    const unsigned long long sT = (unsigned long long)Wd * H;   // 32768
    // layer 1
    gemm_bf16_wmma<<<dim3(B, 2, 2), tb, 0, stream>>>(Xt_bf, wtT[0], S1, nullptr,
        256, 128, 128, 0, sX, 0ull, sT);
    transpose_b2b<<<blocks((long long)B * sT), tb, 0, stream>>>(S1, S2, 256, 128, (long long)B * sT);
    gemm_bf16_wmma<<<dim3(B, 2, 2), tb, 0, stream>>>(Mtt, S2, S3, gt_b[0],
        256, 128, 256, 1, 0ull, sT, sT);
    // layers 2,3
    for (int l = 1; l < 3; ++l) {
        gemm_bf16_wmma<<<dim3(B, 2, 2), tb, 0, stream>>>(S3, wtT[l], S1, nullptr,
            256, 128, 128, 0, sT, 0ull, sT);
        transpose_b2b<<<blocks((long long)B * sT), tb, 0, stream>>>(S1, S2, 256, 128, (long long)B * sT);
        gemm_bf16_wmma<<<dim3(B, 2, 2), tb, 0, stream>>>(Mtt, S2, S3, gt_b[l],
            256, 128, 256, 1, 0ull, sT, sT);
    }
    mean_nodes<<<B, 128, 0, stream>>>(S3, tm, 256, 128);

    // --- projection ---
    proj_kernel<<<B, 64, 0, stream>>>(fm, tm, proj_w, proj_b, outF);
}
